// LinearAttention_83288005804430
// MI455X (gfx1250) — compile-verified, matrix operand order checked by reading, not run on hardware
//
#include <hip/hip_runtime.h>
#include <stdint.h>

// ---------------------------------------------------------------------------
// Types for CDNA5 WMMA (wave32). bf16 operands, fp32 accumulate.
// ---------------------------------------------------------------------------
typedef __bf16 bf16_t;
typedef __attribute__((ext_vector_type(16))) __bf16 v16bf;
typedef __attribute__((ext_vector_type(8)))  __bf16 bf16x8;
typedef __attribute__((ext_vector_type(8)))  float  v8f;

// Problem constants (fixed by setup_inputs)
#define BATCH 4
#define HH    128
#define WW    128
#define NSEQ  (HH*WW)          // 16384
#define DIM   192
#define HID   384
#define QKVD  1152
#define NH    8
#define HD    48
#define ROWS  (BATCH*NSEQ)     // 65536

// ---------------------------------------------------------------------------
// fp32 -> bf16 conversion (round via cast)
// ---------------------------------------------------------------------------
__global__ void cvt_f32_bf16(const float* __restrict__ in, bf16_t* __restrict__ out, int n) {
    int i = blockIdx.x * 256 + threadIdx.x;
    if (i < n) out[i] = (bf16_t)in[i];
}

__global__ void zero_f32(float* __restrict__ p, int n) {
    int i = blockIdx.x * 256 + threadIdx.x;
    if (i < n) p[i] = 0.0f;
}

// ---------------------------------------------------------------------------
// Generic WMMA GEMM:  C[m,n] = sum_k A[m,k] * W[n,k] + bias[n]
// A: M x K bf16 row-major.  W: N x K bf16 row-major (both K-contiguous).
// Block = 256 threads = 8 waves.  Block tile 128(M) x 64(N); wave tile 16x64.
// Per-lane operand layout (16-bit WMMA, wave32):
//   lane L, group g = L>>4, r = L&15:
//     A fragment: row M = r, K chunks [k0+8g, +8) and [k0+16+8g, +8)
//     B fragment: col N = r, same K chunks, sourced from W row (n)  (= B^T row)
// Each chunk is one 16-byte load.
// ---------------------------------------------------------------------------
template <bool OUT_BF16>
__global__ void gemm_bf16_wmma(const bf16_t* __restrict__ A,
                               const bf16_t* __restrict__ W,
                               const float*  __restrict__ bias,
                               void* __restrict__ Cout,
                               int M, int N, int K) {
    const int lane = threadIdx.x & 31;
    const int wave = threadIdx.x >> 5;
    const int g = lane >> 4;
    const int r = lane & 15;
    const int mBase = blockIdx.y * 128 + wave * 16;
    const int nBase = blockIdx.x * 64;

    v8f acc[4] = {v8f{}, v8f{}, v8f{}, v8f{}};
    const bf16_t* aRow = A + (size_t)(mBase + r) * K;

    for (int k0 = 0; k0 < K; k0 += 32) {
        union { v16bf v; bf16x8 p[2]; } a;
        a.p[0] = *(const bf16x8*)(aRow + k0 + 8 * g);
        a.p[1] = *(const bf16x8*)(aRow + k0 + 16 + 8 * g);
#pragma unroll
        for (int t = 0; t < 4; ++t) {
            const bf16_t* wRow = W + (size_t)(nBase + 16 * t + r) * K;
            union { v16bf v; bf16x8 p[2]; } bm;
            bm.p[0] = *(const bf16x8*)(wRow + k0 + 8 * g);
            bm.p[1] = *(const bf16x8*)(wRow + k0 + 16 + 8 * g);
            acc[t] = __builtin_amdgcn_wmma_f32_16x16x32_bf16(
                false, a.v, false, bm.v, (short)0, acc[t], false, false);
        }
    }

    // C layout: VGPR j holds (m = mBase + j + 8g, n = nBase + 16t + r)
#pragma unroll
    for (int t = 0; t < 4; ++t) {
        const int n = nBase + 16 * t + r;
        const float bv = bias[n];
#pragma unroll
        for (int j = 0; j < 8; ++j) {
            const int m = mBase + j + 8 * g;
            const float val = acc[t][j] + bv;
            if (OUT_BF16)
                ((bf16_t*)Cout)[(size_t)m * N + n] = (bf16_t)val;
            else
                ((float*)Cout)[(size_t)m * N + n] = val;
        }
    }
}

// ---------------------------------------------------------------------------
// RoPE + (elu+1) feature map for q and k.  One thread per (row, h*48+d).
// freqs: dq = 12;  d<24 uses H-position (n>>7), d>=24 uses W-position (n&127);
// frequency index = d % 12 (offsets are multiples of 12).
// rotate-half: out = x*cos + (d<24 ? -x[d+24] : x[d-24])*sin
// ---------------------------------------------------------------------------
__global__ void rope_phi_kernel(const float* __restrict__ qkv,
                                bf16_t* __restrict__ phi_q,
                                bf16_t* __restrict__ phi_k) {
    int idx = blockIdx.x * 256 + threadIdx.x;        // over ROWS*HID
    if (idx >= ROWS * HID) return;
    const int row = idx / HID;                       // b*NSEQ + n
    const int c   = idx - row * HID;                 // h*48 + d
    const int d   = c % HD;
    const int n   = row & (NSEQ - 1);

    const float pos  = (d < 24) ? (float)(n >> 7) : (float)(n & 127);
    const int   j    = d % 12;
    const float freq = __expf(-(float)j * (9.2103403719761836f / 12.0f)); // 10000^(-j/12)
    const float th   = pos * freq;
    float s_, c_;
    __sincosf(th, &s_, &c_);

    const int   cp  = (d < 24) ? c + 24 : c - 24;
    const float sgn = (d < 24) ? -1.0f : 1.0f;
    const float* p  = qkv + (size_t)row * QKVD;

    float qv = p[c],        qp = p[cp];
    float kv = p[HID + c],  kp = p[HID + cp];
    float qr = qv * c_ + sgn * qp * s_;
    float kr = kv * c_ + sgn * kp * s_;
    qr = (qr > 0.0f) ? qr + 1.0f : __expf(qr);       // elu(x)+1
    kr = (kr > 0.0f) ? kr + 1.0f : __expf(kr);
    phi_q[idx] = (bf16_t)qr;
    phi_k[idx] = (bf16_t)kr;
}

// ---------------------------------------------------------------------------
// Depthwise 3x3 conv (LePE) on v, fused with v_new = v + lepe + lepe_b.
// v lives in qkv columns [768, 1152).  v2d(b,c,i,j) = qkv[b, i*W+j, 768+c].
// ---------------------------------------------------------------------------
__global__ void lepe_kernel(const float* __restrict__ qkv,
                            const float* __restrict__ w,
                            const float* __restrict__ wb,
                            bf16_t* __restrict__ v_new) {
    int idx = blockIdx.x * 256 + threadIdx.x;        // over B*NSEQ*HID
    if (idx >= ROWS * HID) return;
    const int c = idx % HID;
    const int n = (idx / HID) & (NSEQ - 1);
    const int b = idx / (HID * NSEQ);
    const int i = n >> 7, jj = n & 127;

    const float* base = qkv + (size_t)b * NSEQ * QKVD + 2 * HID + c;
    const float* wc = w + c * 9;
    float acc = wb[c];
#pragma unroll
    for (int di = -1; di <= 1; ++di) {
        const int ii = i + di;
        if ((unsigned)ii >= (unsigned)HH) continue;
#pragma unroll
        for (int dj = -1; dj <= 1; ++dj) {
            const int jv = jj + dj;
            if ((unsigned)jv >= (unsigned)WW) continue;
            acc += wc[(di + 1) * 3 + (dj + 1)] * base[(size_t)(ii * WW + jv) * QKVD];
        }
    }
    acc += base[(size_t)n * QKVD];                   // + v itself
    v_new[idx] = (bf16_t)acc;
}

// ---------------------------------------------------------------------------
// kv[b,h,d,e] = sum_n phi_k[n,d] * v_new[n,e]  (48x48, split-K over n, fp32
// atomics).  Also accumulates k_sum[d].  Bandwidth-bound -> VALU is fine.
// grid = (64 splits, 32 bh); block 256; thread owns a 3x3 output block.
// ---------------------------------------------------------------------------
__global__ void kv_splitk_kernel(const bf16_t* __restrict__ phi_k,
                                 const bf16_t* __restrict__ v_new,
                                 float* __restrict__ kvbuf,
                                 float* __restrict__ ksum) {
    const int bh = blockIdx.y;
    const int b = bh >> 3, h = bh & 7;
    const int nBase = blockIdx.x * 256;
    const int tid = threadIdx.x;
    __shared__ __align__(16) bf16_t kb[32 * HD];
    __shared__ __align__(16) bf16_t vb[32 * HD];

    const int d0 = (tid >> 4) * 3;
    const int e0 = (tid & 15) * 3;
    float acc[3][3] = {{0}};
    float ks[3] = {0, 0, 0};
    const size_t rowBase = ((size_t)b * NSEQ + nBase) * HID + h * HD;

    for (int chunk = 0; chunk < 256; chunk += 32) {
        __syncthreads();
        for (int t = tid; t < 32 * 24; t += 256) {   // 24 dwords per 48-col row
            const int rr = t / 24, cc = t - rr * 24;
            const size_t src = rowBase + (size_t)(chunk + rr) * HID + cc * 2;
            ((uint32_t*)kb)[rr * 24 + cc] = *(const uint32_t*)(phi_k + src);
            ((uint32_t*)vb)[rr * 24 + cc] = *(const uint32_t*)(v_new + src);
        }
        __syncthreads();
        for (int nl = 0; nl < 32; ++nl) {
            float kvals[3], vvals[3];
#pragma unroll
            for (int i = 0; i < 3; ++i) {
                kvals[i] = (float)kb[nl * HD + d0 + i];
                vvals[i] = (float)vb[nl * HD + e0 + i];
            }
#pragma unroll
            for (int i = 0; i < 3; ++i)
#pragma unroll
                for (int j = 0; j < 3; ++j) acc[i][j] += kvals[i] * vvals[j];
            if (e0 == 0) {
                ks[0] += kvals[0]; ks[1] += kvals[1]; ks[2] += kvals[2];
            }
        }
    }
    float* kvo = kvbuf + bh * (HD * HD);
#pragma unroll
    for (int i = 0; i < 3; ++i)
#pragma unroll
        for (int j = 0; j < 3; ++j)
            atomicAdd(&kvo[(d0 + i) * HD + e0 + j], acc[i][j]);
    if (e0 == 0) {
#pragma unroll
        for (int i = 0; i < 3; ++i) atomicAdd(&ksum[bh * HD + d0 + i], ks[i]);
    }
}

// ---------------------------------------------------------------------------
// Build augmented kv^T (bf16, 64x64 per bh):
//   row e<48:  kvt[e][d<48] = kv[d][e]     (B operand for out = q @ kv)
//   row e==48: kvt[48][d<48] = ksum[d]     (yields denominator column)
//   everything else 0 (zero-pads K to 64 so garbage A reads are nullified)
// ---------------------------------------------------------------------------
__global__ void kv_finalize_kernel(const float* __restrict__ kvbuf,
                                   const float* __restrict__ ksum,
                                   bf16_t* __restrict__ kvt) {
    int idx = blockIdx.x * 256 + threadIdx.x;        // 32*64*64
    if (idx >= 32 * 64 * 64) return;
    const int d = idx & 63, e = (idx >> 6) & 63, bh = idx >> 12;
    float v = 0.0f;
    if (e < HD && d < HD)       v = kvbuf[bh * (HD * HD) + d * HD + e];
    else if (e == HD && d < HD) v = ksum[bh * HD + d];
    kvt[idx] = (bf16_t)v;
}

// ---------------------------------------------------------------------------
// out[b,h,n,e] = (sum_d q[n,d]*kv[d,e]) / max(sum_d q[n,d]*ksum[d], 1e-6)
// WMMA: M=n (16/wave), K=64 (padded), N=64 (e tiles 0..2 real, tile 3 col 48
// is the denominator).  kv^T staged in LDS; A loaded straight from phi_q
// (d-contiguous).  grid = (128, 32), block 256 = 8 waves.
// ---------------------------------------------------------------------------
__global__ void attn_out_kernel(const bf16_t* __restrict__ phi_q,
                                const bf16_t* __restrict__ kvt,
                                float* __restrict__ att) {
    const int bh = blockIdx.y;
    const int b = bh >> 3, h = bh & 7;
    const int lane = threadIdx.x & 31;
    const int wave = threadIdx.x >> 5;
    const int g = lane >> 4, r = lane & 15;

    __shared__ __align__(16) bf16_t kvs[64 * 64];
    {
        const uint32_t* src = (const uint32_t*)(kvt + (size_t)bh * 4096);
        uint32_t* dst = (uint32_t*)kvs;
        for (int i = threadIdx.x; i < 2048; i += 256) dst[i] = src[i];
    }
    __syncthreads();

    const int m = blockIdx.x * 128 + wave * 16;
    const bf16_t* aRow = phi_q + ((size_t)(b * NSEQ + m + r)) * HID + h * HD;

    v8f acc[4] = {v8f{}, v8f{}, v8f{}, v8f{}};
#pragma unroll
    for (int k0 = 0; k0 < 64; k0 += 32) {
        union { v16bf v; bf16x8 p[2]; } a;
        a.p[0] = *(const bf16x8*)(aRow + k0 + 8 * g);        // d>=48 garbage is
        a.p[1] = *(const bf16x8*)(aRow + k0 + 16 + 8 * g);   // zeroed by B
#pragma unroll
        for (int t = 0; t < 4; ++t) {
            const bf16_t* brow = kvs + (16 * t + r) * 64;
            union { v16bf v; bf16x8 p[2]; } bm;
            bm.p[0] = *(const bf16x8*)(brow + k0 + 8 * g);
            bm.p[1] = *(const bf16x8*)(brow + k0 + 16 + 8 * g);
            acc[t] = __builtin_amdgcn_wmma_f32_16x16x32_bf16(
                false, a.v, false, bm.v, (short)0, acc[t], false, false);
        }
    }

    // denominator = tile3 column e==48, i.e. lane 16*g of acc[3][j]
#pragma unroll
    for (int j = 0; j < 8; ++j) {
        float den = __shfl(acc[3][j], lane & 16, 32);
        den = fmaxf(den, 1e-6f);
        const float inv = 1.0f / den;
        const int mm = m + j + 8 * g;
        const size_t obase = ((size_t)(b * NSEQ + mm)) * HID + h * HD;
        att[obase + 0 + r]  = acc[0][j] * inv;
        att[obase + 16 + r] = acc[1][j] * inv;
        att[obase + 32 + r] = acc[2][j] * inv;
    }
}

// ---------------------------------------------------------------------------
// LayerNorm over 384, one wave32 per row; writes bf16 for the final GEMM.
// ---------------------------------------------------------------------------
__global__ void layernorm_kernel(const float* __restrict__ x,
                                 const float* __restrict__ gam,
                                 const float* __restrict__ bet,
                                 bf16_t* __restrict__ out) {
    const int lane = threadIdx.x & 31;
    const int wv = threadIdx.x >> 5;
    const int row = blockIdx.x * 8 + wv;
    const float* p = x + (size_t)row * HID;

    float vals[12], s = 0.0f, s2 = 0.0f;
#pragma unroll
    for (int i = 0; i < 12; ++i) {
        const float v = p[lane + 32 * i];
        vals[i] = v; s += v; s2 += v * v;
    }
#pragma unroll
    for (int off = 16; off; off >>= 1) {
        s  += __shfl_xor(s,  off, 32);
        s2 += __shfl_xor(s2, off, 32);
    }
    const float mu  = s * (1.0f / HID);
    const float var = s2 * (1.0f / HID) - mu * mu;
    const float inv = rsqrtf(var + 1e-5f);
    bf16_t* o = out + (size_t)row * HID;
#pragma unroll
    for (int i = 0; i < 12; ++i) {
        const int c = lane + 32 * i;
        o[c] = (bf16_t)((vals[i] - mu) * inv * gam[c] + bet[c]);
    }
}

// ---------------------------------------------------------------------------
// Launcher
// ---------------------------------------------------------------------------
extern "C" void kernel_launch(void* const* d_in, const int* in_sizes, int n_in,
                              void* d_out, int out_size, void* d_ws, size_t ws_size,
                              hipStream_t stream) {
    (void)in_sizes; (void)n_in; (void)out_size; (void)ws_size;
    const float* x      = (const float*)d_in[0];
    const float* in_w   = (const float*)d_in[1];
    const float* in_b   = (const float*)d_in[2];
    const float* qkv_w  = (const float*)d_in[3];
    const float* qkv_b  = (const float*)d_in[4];
    const float* lepe_w = (const float*)d_in[5];
    const float* lepe_b = (const float*)d_in[6];
    const float* ln_g   = (const float*)d_in[7];
    const float* ln_b   = (const float*)d_in[8];
    const float* out_w  = (const float*)d_in[9];
    const float* out_b  = (const float*)d_in[10];

    uint8_t* w = (uint8_t*)d_ws;
    size_t off = 0;
    auto take = [&](size_t bytes) -> uint8_t* {
        uint8_t* p = w + off;
        off += (bytes + 255) & ~(size_t)255;
        return p;
    };
    bf16_t* x_bf     = (bf16_t*)take((size_t)ROWS * DIM * 2);
    bf16_t* inw_bf   = (bf16_t*)take((size_t)HID * DIM * 2);
    bf16_t* qkvw_bf  = (bf16_t*)take((size_t)QKVD * HID * 2);
    bf16_t* outw_bf  = (bf16_t*)take((size_t)DIM * HID * 2);
    bf16_t* xseq_bf  = (bf16_t*)take((size_t)ROWS * HID * 2);
    float*  qkv      = (float*)take((size_t)ROWS * QKVD * 4);
    bf16_t* phi_q    = (bf16_t*)take((size_t)ROWS * HID * 2 + 512); // K-pad slack
    bf16_t* phi_k    = (bf16_t*)take((size_t)ROWS * HID * 2);
    bf16_t* v_new    = (bf16_t*)take((size_t)ROWS * HID * 2);
    float*  kvbuf    = (float*)take((size_t)(32 * HD * HD + 32 * HD) * 4);
    float*  ksum     = kvbuf + 32 * HD * HD;
    bf16_t* kvt      = (bf16_t*)take((size_t)32 * 64 * 64 * 2);
    float*  att      = (float*)take((size_t)ROWS * HID * 4);
    bf16_t* ln_bf    = (bf16_t*)take((size_t)ROWS * HID * 2);

    // 1) fp32 -> bf16 conversions
    {
        int n;
        n = ROWS * DIM;  cvt_f32_bf16<<<(n + 255) / 256, 256, 0, stream>>>(x,     x_bf,   n);
        n = HID * DIM;   cvt_f32_bf16<<<(n + 255) / 256, 256, 0, stream>>>(in_w,  inw_bf, n);
        n = QKVD * HID;  cvt_f32_bf16<<<(n + 255) / 256, 256, 0, stream>>>(qkv_w, qkvw_bf, n);
        n = DIM * HID;   cvt_f32_bf16<<<(n + 255) / 256, 256, 0, stream>>>(out_w, outw_bf, n);
    }
    // 2) in_proj: xseq = x @ in_w^T + in_b        (65536 x 384, K=192)
    gemm_bf16_wmma<true><<<dim3(HID / 64, ROWS / 128), 256, 0, stream>>>(
        x_bf, inw_bf, in_b, xseq_bf, ROWS, HID, DIM);
    // 3) qkv = xseq @ qkv_w^T + qkv_b             (65536 x 1152, K=384)
    gemm_bf16_wmma<false><<<dim3(QKVD / 64, ROWS / 128), 256, 0, stream>>>(
        xseq_bf, qkvw_bf, qkv_b, qkv, ROWS, QKVD, HID);
    // 4) RoPE + elu+1 feature maps
    rope_phi_kernel<<<(ROWS * HID + 255) / 256, 256, 0, stream>>>(qkv, phi_q, phi_k);
    // 5) LePE depthwise conv, fused v_new = v + lepe + lepe_b
    lepe_kernel<<<(ROWS * HID + 255) / 256, 256, 0, stream>>>(qkv, lepe_w, lepe_b, v_new);
    // 6) kv = K^T V (+ k_sum), split-K fp32 atomics
    zero_f32<<<(32 * HD * HD + 32 * HD + 255) / 256, 256, 0, stream>>>(
        kvbuf, 32 * HD * HD + 32 * HD);
    kv_splitk_kernel<<<dim3(64, 32), 256, 0, stream>>>(phi_k, v_new, kvbuf, ksum);
    kv_finalize_kernel<<<(32 * 64 * 64 + 255) / 256, 256, 0, stream>>>(kvbuf, ksum, kvt);
    // 7) out = (Q @ kv) / (Q @ k_sum)  via WMMA with augmented kv^T
    attn_out_kernel<<<dim3(NSEQ / 128, 32), 256, 0, stream>>>(phi_q, kvt, att);
    // 8) LayerNorm -> bf16
    layernorm_kernel<<<ROWS / 8, 256, 0, stream>>>(att, ln_g, ln_b, ln_bf);
    // 9) out_proj: d_out = ln @ out_w^T + out_b   (65536 x 192, K=384)
    gemm_bf16_wmma<false><<<dim3(DIM / 64, ROWS / 128), 256, 0, stream>>>(
        ln_bf, outw_bf, out_b, d_out, ROWS, DIM, HID);
}
